// SecretRQVAE_61718680044222
// MI455X (gfx1250) — matrix operand order repaired and checked by reading, not compiled, not run on hardware
//
#include <hip/hip_runtime.h>
#include <hip/hip_bf16.h>

// ---------------------------------------------------------------------------
// RQ-VAE forward for MI455X (gfx1250, wave32, WMMA bf16, async-to-LDS).
//
// Shapes (from reference):
//   x:        (16,3,256,256)
//   enc1:     4x4 s2 p1, 3->64    -> (16,64,128,128)
//   enc2:     4x4 s2 p1, 64->128  -> (16,128,64,64)
//   enc3:     3x3 s1 p1, 128->256 -> (16,256,64,64)
//   enc4:     3x3 s1 p1, 256->256 -> z (16,256,64,64)
//   RVQ:      N=65536 rows, D=256, K=1024 codes, Q=4 stages
//   dec1:     3x3 s1 p1, 256->128 -> (16,128,64,64)
//   decT1:    4x4 s2 p1, 128->64  -> (16,64,128,128)
//   decT2:    4x4 s2 p1, 64->3    -> recon (16,3,256,256), tanh
// ---------------------------------------------------------------------------

typedef __attribute__((ext_vector_type(16))) __bf16 v16bf;
typedef __attribute__((ext_vector_type(8)))  float  v8f;

union BFrag {
  v16bf  v;
  __bf16 e[16];
  uint4  q[2];
};

__device__ __forceinline__ unsigned int orderKey(float v) {
  unsigned int u = __float_as_uint(v);
  return (u & 0x80000000u) ? ~u : (u | 0x80000000u);
}

// ---------------------------------------------------------------------------
// Codebook prep: f32 -> bf16 copy + per-code squared norms.
// grid: 4096 blocks (Q*K rows), 256 threads (one per D element).
// ---------------------------------------------------------------------------
__global__ void __launch_bounds__(256)
cb_prep_kernel(const float* __restrict__ cb, __bf16* __restrict__ cb_bf,
               float* __restrict__ cnorm) {
  __shared__ float red[256];
  const int row = blockIdx.x;
  const int tid = threadIdx.x;
  const float v = cb[(size_t)row * 256 + tid];
  cb_bf[(size_t)row * 256 + tid] = (__bf16)v;
  red[tid] = v * v;
  __syncthreads();
  for (int s = 128; s > 0; s >>= 1) {
    if (tid < s) red[tid] += red[tid + s];
    __syncthreads();
  }
  if (tid == 0) cnorm[row] = red[0];
}

// ---------------------------------------------------------------------------
// VQ distance GEMM + fused argmin.
// S = R (65536x256, bf16) * CB^T (256x1024, CB stored row-major K x D, bf16)
// score(n,k) = |c_k|^2 - 2 * <r_n, c_k>   (row norm constant per n -> dropped)
// One block = 16-row stripe x all 1024 codes. 8 waves x 8 col tiles each.
// The shared 16x256 A-stripe (8 KB) is DMA'd into LDS once per block with
// global_load_async_to_lds_b128 (ASYNCcnt), then all waves read fragments
// from LDS; the codebook streams through L2 with b128 loads + prefetch.
// ---------------------------------------------------------------------------
__global__ void __launch_bounds__(256)
vq_gemm_argmin_kernel(const __bf16* __restrict__ resid_bf,
                      const __bf16* __restrict__ cb_bf,
                      const float* __restrict__ cnorm,
                      unsigned int* __restrict__ bestIdx) {
  __shared__ __bf16 sA[16 * 256];            // 8 KB A-stripe
  __shared__ unsigned long long sbest[16];

  const int tid = threadIdx.x;
  if (tid < 16) sbest[tid] = 0xFFFFFFFFFFFFFFFFull;

  const int rowBase = blockIdx.x * 16;

  // Async DMA: 8 KB contiguous (16 rows x 512 B) global -> LDS.
  // IOFFSET applies to both LDS and global sides, so one vdst/vaddr pair
  // covers both 4 KB halves. saddr (global base) is workgroup-uniform.
  {
    const __bf16* gbase = resid_bf + (size_t)rowBase * 256;
    unsigned ldsAddr = (unsigned)(uintptr_t)(&sA[0]) + (unsigned)tid * 16u;
    unsigned gOff = (unsigned)tid * 16u;
    asm volatile(
        "global_load_async_to_lds_b128 %0, %1, %2\n\t"
        "global_load_async_to_lds_b128 %0, %1, %2 offset:4096\n\t"
        "s_wait_asynccnt 0x0"
        :
        : "v"(ldsAddr), "v"(gOff), "s"(gbase)
        : "memory");
  }
  __syncthreads();

  const int wave = tid >> 5;
  const int lane = tid & 31;
  const int half = lane >> 4;
  const int ln   = lane & 15;
  const int colBase = wave * 128;

  v8f acc[8] = {};
  const __bf16* aLds = sA + ln * 256;

  for (int kk = 0; kk < 256; kk += 32) {
    // A fragment: 16x32 bf16 from LDS, row = lane%16, K halves by lane/16.
    BFrag a;
    a.q[0] = *(const uint4*)(aLds + kk + half * 16);
    a.q[1] = *(const uint4*)(aLds + kk + half * 16 + 8);
#pragma unroll
    for (int t = 0; t < 8; ++t) {
      const __bf16* crow = cb_bf + (size_t)(colBase + t * 16 + ln) * 256;
      if (kk < 224) __builtin_prefetch((const void*)(crow + kk + 32), 0, 3);
      // B fragment: 32x16 bf16, col = lane%16, contiguous 16 K per lane half.
      BFrag bb;
      bb.q[0] = *(const uint4*)(crow + kk + half * 16);
      bb.q[1] = *(const uint4*)(crow + kk + half * 16 + 8);
      acc[t] = __builtin_amdgcn_wmma_f32_16x16x32_bf16(
          false, a.v, false, bb.v, (short)0, acc[t], false, false);
    }
  }

  // Epilogue: val = |c|^2 - 2*score; pack (orderedVal, idx) -> u64 min in LDS.
#pragma unroll
  for (int t = 0; t < 8; ++t) {
    const int n = colBase + t * 16 + ln;
    const float cn = cnorm[n];
#pragma unroll
    for (int j = 0; j < 8; ++j) {
      const float val = cn - 2.0f * acc[t][j];
      const int m = j + 8 * half;   // C/D layout: VGPR j, lane half -> row
      const unsigned long long key =
          ((unsigned long long)orderKey(val) << 32) | (unsigned long long)n;
      atomicMin(&sbest[m], key);
    }
  }
  __syncthreads();
  if (tid < 16)
    bestIdx[rowBase + tid] = (unsigned int)(sbest[tid] & 0xFFFFFFFFull);
}

// ---------------------------------------------------------------------------
// VQ update: gather winning code, accumulate quantized, update residual
// (f32 + bf16 mirror), accumulate commitment loss, emit index.
// grid: 65536 blocks (one row), 256 threads (one D element).
// ---------------------------------------------------------------------------
__global__ void __launch_bounds__(256)
vq_update_kernel(const unsigned int* __restrict__ bestIdx,
                 const float* __restrict__ cbq,       // stage codebook (1024,256) f32
                 float* __restrict__ resid,
                 __bf16* __restrict__ resid_bf,
                 float* __restrict__ quant,
                 float* __restrict__ lossPtr,         // &commit[q]
                 float* __restrict__ idxOut,          // (B,Q,H,W) region base
                 int q) {
  __shared__ float red[256];
  const int row = blockIdx.x;
  const int d = threadIdx.x;
  const unsigned int idx = bestIdx[row];
  const float c = cbq[(size_t)idx * 256 + d];
  const size_t off = (size_t)row * 256 + d;
  const float r = resid[off];
  const float diff = c - r;
  quant[off] += c;                 // straight-through value == code vector
  const float nr = r - c;
  resid[off] = nr;
  resid_bf[off] = (__bf16)nr;
  red[d] = diff * diff;
  __syncthreads();
  for (int s = 128; s > 0; s >>= 1) {
    if (d < s) red[d] += red[d + s];
    __syncthreads();
  }
  if (d == 0) {
    atomicAdd(lossPtr, red[0] * (1.0f / (65536.0f * 256.0f)));
    const int b = row >> 12;       // / 4096 (H*W)
    const int rem = row & 4095;
    idxOut[((size_t)b * 4 + q) * 4096 + rem] = (float)idx;
  }
}

__global__ void zero_loss_kernel(float* __restrict__ loss) {
  if (threadIdx.x < 4) loss[threadIdx.x] = 0.0f;
}

// ---------------------------------------------------------------------------
// Implicit-GEMM convolution via WMMA bf16.
// Block = 256 threads (8 waves); one 16-output-pixel tile (along W) x all Cout.
// CT = Cout/128 column tiles per wave. Taps looped outside, Cin in 32-steps.
// ---------------------------------------------------------------------------
template <int CT>
__global__ void __launch_bounds__(256)
conv2d_wmma_kernel(const float* __restrict__ in, const float* __restrict__ w,
                   const float* __restrict__ bias, float* __restrict__ out,
                   int B, int Cin, int Hin, int Win,
                   int Cout, int Hout, int Wout,
                   int K, int S, int P, int doRelu) {
  const int tid = threadIdx.x;
  const int wave = tid >> 5;
  const int lane = tid & 31;
  const int half = lane >> 4;
  const int ln   = lane & 15;

  const int wTiles = Wout >> 4;
  int tIdx = blockIdx.x;
  const int xt = tIdx % wTiles; tIdx /= wTiles;
  const int oy = tIdx % Hout;   tIdx /= Hout;
  const int b  = tIdx;

  const int coBase = wave * 16 * CT;
  const int HW = Hin * Win;

  v8f acc[CT] = {};
  const int oxl = xt * 16 + ln;           // A-matrix row for this lane

  for (int ky = 0; ky < K; ++ky) {
    const int iy = oy * S - P + ky;
    if (iy < 0 || iy >= Hin) continue;    // uniform across the block
    for (int kx = 0; kx < K; ++kx) {
      const int ix = oxl * S - P + kx;
      const bool okx = (ix >= 0) && (ix < Win);
      for (int c0 = 0; c0 < Cin; c0 += 32) {
        BFrag a;
#pragma unroll
        for (int e = 0; e < 16; ++e) {
          const int ci = c0 + half * 16 + e;
          const float v = okx
              ? in[(((size_t)b * Cin + ci) * Hin + iy) * Win + ix]
              : 0.0f;
          a.e[e] = (__bf16)v;
        }
#pragma unroll
        for (int t = 0; t < CT; ++t) {
          const int co = coBase + t * 16 + ln;
          BFrag bb;
#pragma unroll
          for (int e = 0; e < 16; ++e) {
            const int ci = c0 + half * 16 + e;
            bb.e[e] = (__bf16)w[((size_t)co * Cin + ci) * (K * K) + ky * K + kx];
          }
          acc[t] = __builtin_amdgcn_wmma_f32_16x16x32_bf16(
              false, a.v, false, bb.v, (short)0, acc[t], false, false);
        }
      }
    }
  }

  const int HWo = Hout * Wout;
#pragma unroll
  for (int t = 0; t < CT; ++t) {
    const int co = coBase + t * 16 + ln;
    const float bv = bias[co];
#pragma unroll
    for (int j = 0; j < 8; ++j) {
      const int m = j + 8 * half;
      const int ox = xt * 16 + m;
      float v = acc[t][j] + bv;
      if (doRelu) v = fmaxf(v, 0.0f);
      out[((size_t)b * Cout + co) * HWo + (size_t)oy * Wout + ox] = v;
    }
  }
}

// ---------------------------------------------------------------------------
// enc1: 4x4 s2 p1, 3->64, ReLU. Direct (C_in too small for WMMA).
// t = b<<20 | co<<14 | oy<<7 | ox.
// ---------------------------------------------------------------------------
__global__ void __launch_bounds__(256)
conv1_kernel(const float* __restrict__ x, const float* __restrict__ w,
             const float* __restrict__ bias, float* __restrict__ out) {
  const int t = blockIdx.x * 256 + threadIdx.x;
  const int ox = t & 127;
  const int oy = (t >> 7) & 127;
  const int co = (t >> 14) & 63;
  const int b  = t >> 20;
  float sum = bias[co];
  for (int ci = 0; ci < 3; ++ci)
    for (int ky = 0; ky < 4; ++ky) {
      const int iy = oy * 2 - 1 + ky;
      if (iy < 0 || iy >= 256) continue;
      for (int kx = 0; kx < 4; ++kx) {
        const int ix = ox * 2 - 1 + kx;
        if (ix < 0 || ix >= 256) continue;
        sum += x[(((size_t)b * 3 + ci) * 256 + iy) * 256 + ix] *
               w[(((size_t)co * 3 + ci) * 4 + ky) * 4 + kx];
      }
    }
  out[t] = fmaxf(sum, 0.0f);
}

// ---------------------------------------------------------------------------
// decT1: ConvTranspose 4x4 s2 p1, 128->64, ReLU. Parity-tap direct form.
// out (16,64,128,128); t = b<<20 | co<<14 | oy<<7 | ox.  w: (128,64,4,4)
// ---------------------------------------------------------------------------
__global__ void __launch_bounds__(256)
convt1_kernel(const float* __restrict__ in, const float* __restrict__ w,
              const float* __restrict__ bias, float* __restrict__ out) {
  const int t = blockIdx.x * 256 + threadIdx.x;
  const int ox = t & 127;
  const int oy = (t >> 7) & 127;
  const int co = (t >> 14) & 63;
  const int b  = t >> 20;
  float sum = bias[co];
  for (int ky = 0; ky < 4; ++ky) {
    const int ty = oy + 1 - ky;
    if (ty < 0 || (ty & 1)) continue;
    const int iy = ty >> 1;
    if (iy >= 64) continue;
    for (int kx = 0; kx < 4; ++kx) {
      const int tx = ox + 1 - kx;
      if (tx < 0 || (tx & 1)) continue;
      const int ix = tx >> 1;
      if (ix >= 64) continue;
      for (int ci = 0; ci < 128; ++ci)
        sum += in[(((size_t)b * 128 + ci) * 64 + iy) * 64 + ix] *
               w[(((size_t)ci * 64 + co) * 4 + ky) * 4 + kx];
    }
  }
  out[t] = fmaxf(sum, 0.0f);
}

// ---------------------------------------------------------------------------
// decT2: ConvTranspose 4x4 s2 p1, 64->3, tanh -> recon.
// out (16,3,256,256); t = ((b*3+co)*256+oy)*256+ox.  w: (64,3,4,4)
// ---------------------------------------------------------------------------
__global__ void __launch_bounds__(256)
convt2_kernel(const float* __restrict__ in, const float* __restrict__ w,
              const float* __restrict__ bias, float* __restrict__ out) {
  const int t = blockIdx.x * 256 + threadIdx.x;
  const int ox = t & 255;
  const int oy = (t >> 8) & 255;
  const int cb = t >> 16;
  const int co = cb % 3;
  const int b  = cb / 3;
  float sum = bias[co];
  for (int ky = 0; ky < 4; ++ky) {
    const int ty = oy + 1 - ky;
    if (ty < 0 || (ty & 1)) continue;
    const int iy = ty >> 1;
    if (iy >= 128) continue;
    for (int kx = 0; kx < 4; ++kx) {
      const int tx = ox + 1 - kx;
      if (tx < 0 || (tx & 1)) continue;
      const int ix = tx >> 1;
      if (ix >= 128) continue;
      for (int ci = 0; ci < 64; ++ci)
        sum += in[(((size_t)b * 64 + ci) * 128 + iy) * 128 + ix] *
               w[(((size_t)ci * 3 + co) * 4 + ky) * 4 + kx];
    }
  }
  out[t] = tanhf(sum);
}

// ---------------------------------------------------------------------------
// NCHW z -> (N,D) residual rows (+bf16 mirror, zero quantized accumulator).
// 32x32 LDS tile transpose per batch. grid (128,8,16), block (32,8).
// ---------------------------------------------------------------------------
__global__ void chw_to_nd_kernel(const float* __restrict__ zchw,
                                 float* __restrict__ resid,
                                 __bf16* __restrict__ resid_bf,
                                 float* __restrict__ quant) {
  __shared__ float tile[32][33];
  const int b = blockIdx.z;
  const int sBase = blockIdx.x * 32;
  const int dBase = blockIdx.y * 32;
  const int tx = threadIdx.x, ty = threadIdx.y;
#pragma unroll
  for (int i = 0; i < 4; ++i) {
    const int d = dBase + ty + i * 8;
    tile[ty + i * 8][tx] = zchw[((size_t)b * 256 + d) * 4096 + sBase + tx];
  }
  __syncthreads();
#pragma unroll
  for (int i = 0; i < 4; ++i) {
    const int s = sBase + ty + i * 8;
    const int d = dBase + tx;
    const float v = tile[tx][ty + i * 8];
    const size_t off = ((size_t)b * 4096 + s) * 256 + d;
    resid[off] = v;
    resid_bf[off] = (__bf16)v;
    quant[off] = 0.0f;
  }
}

// (N,D) quantized rows -> NCHW q_chw output.
__global__ void nd_to_chw_kernel(const float* __restrict__ quant,
                                 float* __restrict__ qchw) {
  __shared__ float tile[32][33];
  const int b = blockIdx.z;
  const int sBase = blockIdx.x * 32;
  const int dBase = blockIdx.y * 32;
  const int tx = threadIdx.x, ty = threadIdx.y;
#pragma unroll
  for (int i = 0; i < 4; ++i) {
    const int s = sBase + ty + i * 8;
    tile[ty + i * 8][tx] = quant[((size_t)b * 4096 + s) * 256 + dBase + tx];
  }
  __syncthreads();
#pragma unroll
  for (int i = 0; i < 4; ++i) {
    const int d = dBase + ty + i * 8;
    const int s = sBase + tx;
    qchw[((size_t)b * 256 + d) * 4096 + s] = tile[tx][ty + i * 8];
  }
}

// ---------------------------------------------------------------------------
extern "C" void kernel_launch(void* const* d_in, const int* in_sizes, int n_in,
                              void* d_out, int out_size, void* d_ws, size_t ws_size,
                              hipStream_t stream) {
  (void)in_sizes; (void)n_in; (void)out_size; (void)ws_size;

  const float* x    = (const float*)d_in[0];
  const float* ew1  = (const float*)d_in[1];
  const float* eb1  = (const float*)d_in[2];
  const float* ew2  = (const float*)d_in[3];
  const float* eb2  = (const float*)d_in[4];
  const float* ew3  = (const float*)d_in[5];
  const float* eb3  = (const float*)d_in[6];
  const float* ew4  = (const float*)d_in[7];
  const float* eb4  = (const float*)d_in[8];
  const float* cbks = (const float*)d_in[9];
  const float* dw1  = (const float*)d_in[10];
  const float* db1  = (const float*)d_in[11];
  const float* tw1  = (const float*)d_in[12];
  const float* tb1  = (const float*)d_in[13];
  const float* tw2  = (const float*)d_in[14];
  const float* tb2  = (const float*)d_in[15];

  // Workspace layout (bytes); buffers reused once their producer/consumer
  // pairs retire (stream-ordered).
  char* ws = (char*)d_ws;
  float*        A1 = (float*)(ws);                        // 64 MB: enc1 out, then residual rows
  float*        A2 = (float*)(ws + 67108864);             // 32 MB: enc2 out, then dec1 out
  float*        A3 = (float*)(ws + 100663296);            // 64 MB: enc3 out, then quantized rows
  float*        ZC = (float*)(ws + 167772160);            // 64 MB: enc4(z) out, then decT1 out
  __bf16*       RB = (__bf16*)(ws + 234881024);           // 32 MB: residual bf16 mirror
  __bf16*       CB = (__bf16*)(ws + 268435456);           //  2 MB: codebooks bf16
  float*        CN = (float*)(ws + 270532608);            // 16 KB: code norms
  unsigned int* BI = (unsigned int*)(ws + 270548992);     // 256 KB: per-row winners

  float* RS = A1;   // residual rows f32
  float* QT = A3;   // quantized rows f32

  float* out   = (float*)d_out;
  float* RECON = out;                 // 16*3*256*256   = 3145728
  float* IDX   = out + 3145728;       // 16*4*64*64     = 262144
  float* LOSS  = out + 3407872;       // 4
  float* QCHW  = out + 3407876;       // 16*256*64*64   = 16777216

  // Encoder
  cb_prep_kernel<<<4096, 256, 0, stream>>>(cbks, CB, CN);
  conv1_kernel<<<65536, 256, 0, stream>>>(x, ew1, eb1, A1);
  conv2d_wmma_kernel<1><<<4096, 256, 0, stream>>>(A1, ew2, eb2, A2,
      16, 64, 128, 128, 128, 64, 64, 4, 2, 1, 1);
  conv2d_wmma_kernel<2><<<4096, 256, 0, stream>>>(A2, ew3, eb3, A3,
      16, 128, 64, 64, 256, 64, 64, 3, 1, 1, 1);
  conv2d_wmma_kernel<2><<<4096, 256, 0, stream>>>(A3, ew4, eb4, ZC,
      16, 256, 64, 64, 256, 64, 64, 3, 1, 1, 0);

  // Residual VQ
  chw_to_nd_kernel<<<dim3(128, 8, 16), dim3(32, 8), 0, stream>>>(ZC, RS, RB, QT);
  zero_loss_kernel<<<1, 32, 0, stream>>>(LOSS);
  for (int q = 0; q < 4; ++q) {
    vq_gemm_argmin_kernel<<<4096, 256, 0, stream>>>(
        RB, CB + (size_t)q * 1024 * 256, CN + q * 1024, BI);
    vq_update_kernel<<<65536, 256, 0, stream>>>(
        BI, cbks + (size_t)q * 1024 * 256, RS, RB, QT, LOSS + q, IDX, q);
  }
  nd_to_chw_kernel<<<dim3(128, 8, 16), dim3(32, 8), 0, stream>>>(QT, QCHW);

  // Decoder
  conv2d_wmma_kernel<1><<<4096, 256, 0, stream>>>(QCHW, dw1, db1, A2,
      16, 256, 64, 64, 128, 64, 64, 3, 1, 1, 1);
  convt1_kernel<<<65536, 256, 0, stream>>>(A2, tw1, tb1, ZC);
  convt2_kernel<<<12288, 256, 0, stream>>>(ZC, tw2, tb2, RECON);
}
